// BCM_Conv2d_fft_75754633167433
// MI455X (gfx1250) — matrix operand order repaired and checked by reading, not compiled, not run on hardware
//
#include <hip/hip_runtime.h>
#include <hip/hip_bf16.h>

typedef __attribute__((ext_vector_type(16))) _Float16 v16h;
typedef __attribute__((ext_vector_type(8)))  _Float16 v8h;
typedef __attribute__((ext_vector_type(2)))  _Float16 v2h;
typedef __attribute__((ext_vector_type(2)))  __fp16   v2fp;
typedef __attribute__((ext_vector_type(8)))  float    v8f;

#define CIN   256
#define COUT  256
#define HIMG  56
#define WIMG  56
#define HWIMG 3136              // 56*56
#define KDIM  2304              // 9*256
#define NTOT  25088             // 8*3136
#define BM    128
#define BN    128
#define BK    32
#define ASTR  40                // halves per As row (80B, 16B aligned)
#define BSTR  40                // halves per Bs row

// ---------------------------------------------------------------------------
// Kernel 1: expand circulant blocks w[32][288][8] -> dense W_eff[256][2304] f16
//   W_eff[p*8+j][k*8+t] = w[p][k][(j - t) & 7]
// ---------------------------------------------------------------------------
__global__ __launch_bounds__(256) void bcm_expand_w(const float* __restrict__ w,
                                                    _Float16* __restrict__ weff) {
    int idx = blockIdx.x * blockDim.x + threadIdx.x;   // 256*2304 total
    if (idx >= COUT * KDIM) return;
    int row = idx / KDIM;          // output channel
    int col = idx - row * KDIM;    // im2col row index (kh,kw,c order)
    int p = row >> 3, j = row & 7;
    int k = col >> 3, t = col & 7;
    float v = w[(p * 288 + k) * 8 + ((j - t) & 7)];
    weff[idx] = (_Float16)v;
}

// ---------------------------------------------------------------------------
// Pipeline stage: one K-chunk of registers (A: 16 halves, B: 16 floats + mask)
// ---------------------------------------------------------------------------
struct Stage {
    v8h   a0, a1;
    float b[16];
    float ok;
};

// Unconditional (clamped-address) im2col fetch of 16 channels for one pixel.
// The zero-padding mask is a 0/1 float applied after the loads return, so all
// 16 global_load_b32 issue as one clause with a single (partial) wait.
__device__ __forceinline__ void fetch_b(const float* __restrict__ xb,
                                        int h0, int w0, int khi, int kc,
                                        Stage& s) {
    const int khkw = kc >> 8;                 // 0..8, scalar
    const int q    = khkw / 3;
    const int dh   = q - 1;
    const int dw   = khkw - q * 3 - 1;
    const int ph   = h0 + dh, pw = w0 + dw;
    s.ok = (((unsigned)ph < (unsigned)HIMG) & ((unsigned)pw < (unsigned)WIMG))
               ? 1.0f : 0.0f;
    const int phc = min(max(ph, 0), HIMG - 1);
    const int pwc = min(max(pw, 0), WIMG - 1);
    const float* xp = xb + (size_t)((kc & 255) + khi * 16) * HWIMG
                         + phc * WIMG + pwc;
    #pragma unroll
    for (int i = 0; i < 16; ++i)
        s.b[i] = xp[(size_t)i * HWIMG];
}

// ---------------------------------------------------------------------------
// Kernel 2: implicit-im2col GEMM  Out[256][25088] = W_eff[256][2304] x X + b
// 256 threads (8 waves), 128x128 tile, K-chunks of 32, f16 WMMA / f32 accum.
// Software-pipelined with ping-pong stages (K loop unrolled by 2 -> no
// register rotation copies).
// ---------------------------------------------------------------------------
__global__ __launch_bounds__(256) void bcm_gemm_wmma(const float* __restrict__ x,
                                                     const _Float16* __restrict__ weff,
                                                     const float* __restrict__ bias,
                                                     float* __restrict__ out) {
    __shared__ _Float16 As[BM * ASTR];   // [M=128][K=32] (+pad)
    __shared__ _Float16 Bs[BN * BSTR];   // [N=128][K=32] (+pad)

    const int tid  = threadIdx.x;
    const int lane = tid & 31;
    const int wave = tid >> 5;           // 0..7
    const int wm   = wave >> 1;          // 0..3  -> 32 M rows
    const int wn   = wave & 1;           // 0..1  -> 64 N cols

    const int m0 = blockIdx.y * BM;
    const int n0 = blockIdx.x * BN;

    // --- B-load mapping: each thread owns one spatial position, 16 channels ---
    const int nn   = tid & 127;                   // N position within tile
    const int khi  = tid >> 7;                    // 0/1 -> K half within chunk
    const int n    = n0 + nn;
    const int bimg = n / HWIMG;
    const int hw   = n - bimg * HWIMG;
    const int h0   = hw / WIMG;
    const int w0   = hw - h0 * WIMG;
    const float* xb = x + (size_t)bimg * CIN * HWIMG;

    // --- A-load mapping: 2 threads per W_eff row, 16 halves each ---
    const int arow  = tid >> 1;                   // 0..127
    const int ahalf = (tid & 1) * 16;             // 0 / 16 halves
    const _Float16* wsrc = weff + (size_t)(m0 + arow) * KDIM + ahalf;
    _Float16* adst = &As[arow * ASTR + ahalf];
    _Float16* brow = &Bs[nn * BSTR + khi * 16];

    v8f acc[2][4];
    #pragma unroll
    for (int i = 0; i < 2; ++i)
        #pragma unroll
        for (int j = 0; j < 4; ++j)
            acc[i][j] = v8f{};

    const int mlane = lane & 15;
    const int half  = lane >> 4;         // 0/1
    const int akoff = half * 16;         // A frag: K 0-15 / 16-31
    const int bkoff = half * 16;         // B frag: K 0-15 / 16-31

    // One pipeline step: commit `cur` to LDS, prefetch kc+BK into `nxt`,
    // run the 8 WMMAs for `cur`'s chunk.
    auto step = [&](Stage& cur, Stage& nxt, int kc) {
        // ---- commit current chunk to LDS ----
        ((v8h*)adst)[0] = cur.a0;
        ((v8h*)adst)[1] = cur.a1;
        #pragma unroll
        for (int i = 0; i < 8; ++i) {
            v2fp pkr = __builtin_amdgcn_cvt_pkrtz(cur.b[2 * i] * cur.ok,
                                                  cur.b[2 * i + 1] * cur.ok);
            *(v2h*)(brow + 2 * i) = __builtin_bit_cast(v2h, pkr);
        }
        __syncthreads();

        // ---- prefetch next chunk (overlaps the WMMAs below) ----
        const int kn = kc + BK;
        if (kn < KDIM) {
            const v8h* s = (const v8h*)(wsrc + kn);
            nxt.a0 = s[0]; nxt.a1 = s[1];
            fetch_b(xb, h0, w0, khi, kn, nxt);
        }

        // ---- load all fragments, then 8 back-to-back WMMAs ----
        v16h afrag[2], bfrag[4];
        #pragma unroll
        for (int mt = 0; mt < 2; ++mt) {
            const _Float16* ap = &As[(wm * 32 + mt * 16 + mlane) * ASTR];
            v8h lo = *(const v8h*)(ap + akoff);
            v8h hi = *(const v8h*)(ap + 8 + akoff);
            afrag[mt] = __builtin_shufflevector(lo, hi,
                0,1,2,3,4,5,6,7,8,9,10,11,12,13,14,15);
        }
        #pragma unroll
        for (int nt = 0; nt < 4; ++nt) {
            const _Float16* bp = &Bs[(wn * 64 + nt * 16 + mlane) * BSTR + bkoff];
            v8h lo = *(const v8h*)(bp);
            v8h hi = *(const v8h*)(bp + 8);
            bfrag[nt] = __builtin_shufflevector(lo, hi,
                0,1,2,3,4,5,6,7,8,9,10,11,12,13,14,15);
        }
        #pragma unroll
        for (int nt = 0; nt < 4; ++nt)
            #pragma unroll
            for (int mt = 0; mt < 2; ++mt)
                acc[mt][nt] = __builtin_amdgcn_wmma_f32_16x16x32_f16(
                    false, afrag[mt], false, bfrag[nt],
                    (short)0, acc[mt][nt], false, false);
        __syncthreads();
    };

    // ---- pipeline prologue: fetch chunk 0 into stage 0 ----
    Stage st0 = {}, st1 = {};
    {
        const v8h* s = (const v8h*)wsrc;
        st0.a0 = s[0]; st0.a1 = s[1];
    }
    fetch_b(xb, h0, w0, khi, 0, st0);

    // ---- main loop: 72 chunks, unrolled 2x with ping-pong stages ----
    for (int kc = 0; kc < KDIM; kc += 2 * BK) {
        step(st0, st1, kc);
        step(st1, st0, kc + BK);
    }

    // ---- epilogue: C/D layout -> [B, COUT, H, W] f32 + bias ----
    #pragma unroll
    for (int nt = 0; nt < 4; ++nt) {
        const int ncol = n0 + wn * 64 + nt * 16 + mlane;
        const int bo   = ncol / HWIMG;
        const int hwo  = ncol - bo * HWIMG;
        #pragma unroll
        for (int mt = 0; mt < 2; ++mt) {
            #pragma unroll
            for (int r = 0; r < 8; ++r) {
                const int m = m0 + wm * 32 + mt * 16 + half * 8 + r;
                out[((size_t)(bo * COUT + m)) * HWIMG + hwo] = acc[mt][nt][r] + bias[m];
            }
        }
    }
}

// ---------------------------------------------------------------------------
extern "C" void kernel_launch(void* const* d_in, const int* in_sizes, int n_in,
                              void* d_out, int out_size, void* d_ws, size_t ws_size,
                              hipStream_t stream) {
    (void)in_sizes; (void)n_in; (void)out_size; (void)ws_size;
    const float* x = (const float*)d_in[0];   // [8,256,56,56]
    const float* w = (const float*)d_in[1];   // [32,288,8]
    const float* b = (const float*)d_in[2];   // [256]
    float* out = (float*)d_out;               // [8,256,56,56]

    _Float16* weff = (_Float16*)d_ws;         // [256][2304] f16 = 1.125 MB

    // 1) expand circulant blocks to dense f16 weight matrix
    bcm_expand_w<<<(COUT * KDIM) / 256, 256, 0, stream>>>(w, weff);

    // 2) implicit-im2col WMMA GEMM: grid = (25088/128, 256/128)
    dim3 grid(NTOT / BN, COUT / BM);
    bcm_gemm_wmma<<<grid, 256, 0, stream>>>(x, weff, b, out);
}
